// MaskPredictor_62654982914469
// MI455X (gfx1250) — matrix operand order repaired and checked by reading, not compiled, not run on hardware
//
#include <hip/hip_runtime.h>
#include <hip/hip_bf16.h>
#include <cstdint>
#include <cstddef>

// ---------------------------------------------------------------------------
// Types for CDNA5 WMMA (wave32): bf16 operands, f32 accumulation.
// ---------------------------------------------------------------------------
typedef __bf16 bf16_t;
typedef __attribute__((ext_vector_type(16))) __bf16 v16bf;
typedef __attribute__((ext_vector_type(8)))  float  v8f;

#define BM 128           // block tile M
#define BN 64            // block tile N (Cout); all Cout are multiples of 64
#define BK 32            // K step = WMMA bf16 K
#define LDA_P 48         // padded LDS stride (elems) -> 96B rows, 16B aligned
#define LDB_P 48

// CDNA5 async global->LDS copy (ASYNCcnt-tracked, no VGPR staging).
// IOFFSET applies to both the LDS and the global address.
#define ASYNC_LDS_B128_X2(ldsoff, gptr)                                       \
  asm volatile("global_load_async_to_lds_b128 %0, %1, off\n\t"                \
               "global_load_async_to_lds_b128 %0, %1, off offset:16"          \
               :: "v"(ldsoff), "v"(gptr) : "memory")
#define ASYNC_LDS_B128(ldsoff, gptr)                                          \
  asm volatile("global_load_async_to_lds_b128 %0, %1, off"                    \
               :: "v"(ldsoff), "v"(gptr) : "memory")
#define WAIT_ASYNC0()                                                         \
  asm volatile("s_wait_asynccnt 0x0" ::: "memory")

// Flat pointer -> wave-relative LDS byte offset (flat-LDS aperture keeps the
// offset in addr[31:0] per the CDNA5 aperture rules).
__device__ __forceinline__ unsigned lds_off(const void* p) {
  return (unsigned)(uintptr_t)p;
}

// ---------------------------------------------------------------------------
// Implicit-GEMM conv (and FC) with fused BN / residual / ReLU / GELU epilogue.
// Double-buffered LDS, async-copy pipelined: copies for K-tile t+1 fly while
// WMMAs chew on tile t.
// in  : NHWC bf16 [Nimg, H, W, Cin]
// wt  : bf16 [Cout, Kpad], K ordered (kh, kw, cin), zero-padded to Kpad
// out : NHWC bf16 [M, Cout], M = Nimg*Ho*Wo
// res : optional residual, same layout as out
// flags: bit0 = ReLU, bit1 = exact GELU
// ---------------------------------------------------------------------------
__global__ __launch_bounds__(256)
void gemm_conv_bf16_wmma(const bf16_t* __restrict__ in,
                         const bf16_t* __restrict__ wt,
                         bf16_t* __restrict__ out,
                         const float* __restrict__ scale,
                         const float* __restrict__ shift,
                         const bf16_t* __restrict__ res,
                         int H, int W, int Cin,
                         int Ho, int Wo, int Cout,
                         int KH, int KW, int stride, int pad,
                         int Ktot, int Kpad, int M, int flags)
{
  __shared__ alignas(16) bf16_t As[2][BM * LDA_P];
  __shared__ alignas(16) bf16_t Bs[2][BN * LDB_P];

  const int tid  = threadIdx.x;
  const int lane = tid & 31;
  const int wid  = tid >> 5;        // 8 waves
  const int wm   = wid >> 1;        // 0..3  (M direction)
  const int wn   = wid & 1;         // 0..1  (N direction)
  const int half = lane >> 4;       // lane half (ISA fragment layout)
  const int l16  = lane & 15;

  // ---- A (im2col) global-load mapping: 2 threads per row, 16 k-elems each
  const int ar    = tid >> 1;             // 0..127
  const int akseg = (tid & 1) * 16;       // 0 or 16
  const int am    = blockIdx.x * BM + ar;
  int img = 0, oh = 0, ow = 0;
  if (am < M) {
    img = am / (Ho * Wo);
    int r2 = am % (Ho * Wo);
    oh = r2 / Wo; ow = r2 % Wo;
  }

  // ---- B (weights) global-load mapping: 4 threads per row, 8 k-elems each
  const int bnr   = tid >> 2;             // 0..63
  const int bkseg = (tid & 3) * 8;
  const bf16_t* wrow = wt + (size_t)(blockIdx.y * BN + bnr) * Kpad + bkseg;

  const bool chanfast = ((Cin & 15) == 0); // channel-contiguous vector path

  // Per-buffer LDS byte offsets for the async-copy destinations.
  const unsigned aoffs[2] = { lds_off(&As[0][ar * LDA_P + akseg]),
                              lds_off(&As[1][ar * LDA_P + akseg]) };
  const unsigned boffs[2] = { lds_off(&Bs[0][bnr * LDB_P + bkseg]),
                              lds_off(&Bs[1][bnr * LDB_P + bkseg]) };

  // Stage one BK-wide K-tile (tile start kt) into LDS buffer bufi.
  auto stage = [&](int kt, int bufi) {
    if (chanfast) {
      const int kbase = kt + akseg;        // multiple of 16
      bool ok = false;
      const bf16_t* src = in;
      if (am < M && kbase < Ktot) {
        int p  = kbase / Cin;              // (kh,kw) patch index
        int c0 = kbase % Cin;              // 16-aligned channel base
        int kh = p / KW, kw = p % KW;
        int ih = oh * stride - pad + kh;
        int iw = ow * stride - pad + kw;
        if (ih >= 0 && ih < H && iw >= 0 && iw < W) {
          ok = true;
          src = in + (((size_t)img * H + ih) * W + iw) * Cin + c0;
        }
      }
      if (ok) {
        ASYNC_LDS_B128_X2(aoffs[bufi], src); // 32B: memory -> LDS direct
      } else {
        const uint4 z = make_uint4(0u, 0u, 0u, 0u);
        uint4* dst = (uint4*)(&As[bufi][ar * LDA_P + akseg]);
        dst[0] = z; dst[1] = z;            // conv padding / M tail: zero fill
      }
    } else {                               // stem: Cin == 3, scalar gather
      #pragma unroll 4
      for (int j = 0; j < 16; ++j) {
        int k = kt + akseg + j;
        bf16_t val = (bf16_t)0.0f;
        if (am < M && k < Ktot) {
          int c = k % Cin, p = k / Cin;
          int kh = p / KW, kw = p % KW;
          int ih = oh * stride - pad + kh;
          int iw = ow * stride - pad + kw;
          if (ih >= 0 && ih < H && iw >= 0 && iw < W)
            val = in[(((size_t)img * H + ih) * W + iw) * Cin + c];
        }
        As[bufi][ar * LDA_P + akseg + j] = val;
      }
    }
    // Weights tile: always in-bounds (Cout % 64 == 0, Kpad % 32 == 0).
    ASYNC_LDS_B128(boffs[bufi], wrow + kt);
  };

  v8f acc[2][2] = {};
  const int nT = Kpad / BK;

  stage(0, 0);                             // prologue: preload tile 0

  for (int t = 0; t < nT; ++t) {
    const int cb = t & 1;
    WAIT_ASYNC0();                         // tile t's async copies done
    __syncthreads();                       // tile t published to all waves

    if (t + 1 < nT) stage((t + 1) * BK, cb ^ 1);  // overlap next copies

    const bf16_t* Ab = &As[cb][0];
    const bf16_t* Bb = &Bs[cb][0];

    v16bf afrag[2], bfrag[2];
    #pragma unroll
    for (int sm = 0; sm < 2; ++sm) {
      // A 16x32 bf16 layout: lane half selects K+8; VGPRs 0..3 -> K 0..7(+8),
      // VGPRs 4..7 -> K 16..23(+8): two contiguous 8-element K runs.
      const bf16_t* ab = Ab + (wm * 32 + sm * 16 + l16) * LDA_P + half * 8;
      ((uint4*)&afrag[sm])[0] = *(const uint4*)(ab);
      ((uint4*)&afrag[sm])[1] = *(const uint4*)(ab + 16);
    }
    #pragma unroll
    for (int sn = 0; sn < 2; ++sn) {
      // B 32x16 bf16 layout: same K order as A: lane half selects K+8; elems
      // 0..7 -> K 0..7(+8), elems 8..15 -> K 16..23(+8) (Bs is stored [n][k]).
      const bf16_t* bb = Bb + (wn * 32 + sn * 16 + l16) * LDB_P + half * 8;
      ((uint4*)&bfrag[sn])[0] = *(const uint4*)(bb);
      ((uint4*)&bfrag[sn])[1] = *(const uint4*)(bb + 16);
    }
    #pragma unroll
    for (int sm = 0; sm < 2; ++sm)
      #pragma unroll
      for (int sn = 0; sn < 2; ++sn)
        acc[sm][sn] = __builtin_amdgcn_wmma_f32_16x16x32_bf16(
            false, afrag[sm], false, bfrag[sn], (short)0, acc[sm][sn],
            false, false);

    __syncthreads();                       // reads of buffer cb complete
  }

  // ------------ epilogue: BN fold, residual, activation ------------
  #pragma unroll
  for (int sn = 0; sn < 2; ++sn) {
    const int ch = blockIdx.y * BN + wn * 32 + sn * 16 + l16;
    const float sc = scale[ch], sh = shift[ch];
    #pragma unroll
    for (int sm = 0; sm < 2; ++sm) {
      #pragma unroll
      for (int r = 0; r < 8; ++r) {
        const int mrow = blockIdx.x * BM + wm * 32 + sm * 16 + half * 8 + r;
        if (mrow < M) {
          float v = acc[sm][sn][r] * sc + sh;
          if (res) v += (float)res[(size_t)mrow * Cout + ch];
          if (flags & 1) v = fmaxf(v, 0.0f);
          if (flags & 2) v = 0.5f * v * (1.0f + erff(v * 0.70710678118654752f));
          out[(size_t)mrow * Cout + ch] = (bf16_t)v;
        }
      }
    }
  }
}

// ---------------------------------------------------------------------------
// Mask-multiply + channel-group reshape: fp32 NCHW -> bf16 NHWC [64,256,256,3]
// ---------------------------------------------------------------------------
__global__ void stem_prep(const float* __restrict__ x,
                          const int* __restrict__ masks,
                          bf16_t* __restrict__ out, int total)
{
  int idx = blockIdx.x * blockDim.x + threadIdx.x;
  if (idx >= total) return;
  int cc = idx % 3;  int t = idx / 3;
  int w  = t % 256;  t /= 256;
  int h  = t % 256;  t /= 256;
  int im = t;                       // 0..63
  int n = im >> 5, g = im & 31;
  int c = g * 3 + cc;
  int win = c / 24;
  float mv = (float)masks[(((size_t)n * 4 + win) * 256 + h) * 256 + w];
  float xv = x[(((size_t)n * 96 + c) * 256 + h) * 256 + w];
  out[idx] = (bf16_t)(xv * mv);
}

// ---------------------------------------------------------------------------
// Weight repack: OIHW fp32 -> [O][Kpad] bf16, K ordered (kh,kw,ci), zero pad.
// Also handles FC weights with KH=KW=1.
// ---------------------------------------------------------------------------
__global__ void repack_w(const float* __restrict__ src, bf16_t* __restrict__ dst,
                         int I, int KH, int KW, int Ktot, int Kpad, int total)
{
  int idx = blockIdx.x * blockDim.x + threadIdx.x;
  if (idx >= total) return;
  int k = idx % Kpad, o = idx / Kpad;
  float v = 0.0f;
  if (k < Ktot) {
    int ci = k % I, p = k / I;
    int kh = p / KW, kw = p % KW;
    v = src[(((size_t)o * I + ci) * KH + kh) * KW + kw];
  }
  dst[idx] = (bf16_t)v;
}

// ---------------------------------------------------------------------------
// BN fold: scale = g*rsqrt(v+eps); shift = b - m*scale (+ bias*scale for FC)
// ---------------------------------------------------------------------------
__global__ void bn_fold(const float* __restrict__ g, const float* __restrict__ b,
                        const float* __restrict__ m, const float* __restrict__ v,
                        const float* __restrict__ bias,
                        float* __restrict__ scale, float* __restrict__ shift, int C)
{
  int c = blockIdx.x * blockDim.x + threadIdx.x;
  if (c >= C) return;
  float sc = g[c] * rsqrtf(v[c] + 1e-5f);
  float sh = b[c] - m[c] * sc;
  if (bias) sh += bias[c] * sc;
  scale[c] = sc; shift[c] = sh;
}

// ---------------------------------------------------------------------------
// 3x3 stride-2 pad-1 maxpool, NHWC bf16
// ---------------------------------------------------------------------------
__global__ void maxpool3x3s2(const bf16_t* __restrict__ in, bf16_t* __restrict__ out,
                             int Hi, int Wi, int Ho, int Wo, int C, int total)
{
  int idx = blockIdx.x * blockDim.x + threadIdx.x;
  if (idx >= total) return;
  int c  = idx % C;  int t = idx / C;
  int ow = t % Wo;   t /= Wo;
  int oh = t % Ho;   t /= Ho;
  int img = t;
  float best = -3.0e38f;
  #pragma unroll
  for (int kh = 0; kh < 3; ++kh) {
    int ih = oh * 2 - 1 + kh;
    if (ih < 0 || ih >= Hi) continue;
    #pragma unroll
    for (int kw = 0; kw < 3; ++kw) {
      int iw = ow * 2 - 1 + kw;
      if (iw < 0 || iw >= Wi) continue;
      best = fmaxf(best, (float)in[(((size_t)img * Hi + ih) * Wi + iw) * C + c]);
    }
  }
  out[idx] = (bf16_t)best;
}

// Global average pool, NHWC bf16 [Nimg, HW, C] -> [Nimg, C]
__global__ void avgpool_global(const bf16_t* __restrict__ in, bf16_t* __restrict__ out,
                               int HW, int C, int total)
{
  int idx = blockIdx.x * blockDim.x + threadIdx.x;
  if (idx >= total) return;
  int c = idx % C, img = idx / C;
  float s = 0.0f;
  for (int p = 0; p < HW; ++p)
    s += (float)in[((size_t)img * HW + p) * C + c];
  out[idx] = (bf16_t)(s / (float)HW);
}

// [2,32,64] mean over groups -> fp32 [2,64]
__global__ void final_mean(const bf16_t* __restrict__ in, float* __restrict__ out,
                           int total)
{
  int idx = blockIdx.x * blockDim.x + threadIdx.x;
  if (idx >= total) return;
  int j = idx % 64, b = idx / 64;
  float s = 0.0f;
  #pragma unroll 8
  for (int g = 0; g < 32; ++g)
    s += (float)in[(size_t)(b * 32 + g) * 64 + j];
  out[idx] = s * (1.0f / 32.0f);
}

// ---------------------------------------------------------------------------
// Host orchestration
// ---------------------------------------------------------------------------
struct BNP { const float *g, *b, *m, *v; };
struct BlkP { const float* w1; BNP bn1; const float* w2; BNP bn2;
              bool has_down; const float* dw; BNP dbn; };
struct FcP { const float *w, *b; BNP bn; };

extern "C" void kernel_launch(void* const* d_in, const int* in_sizes, int n_in,
                              void* d_out, int out_size, void* d_ws, size_t ws_size,
                              hipStream_t stream)
{
  (void)in_sizes; (void)n_in; (void)out_size; (void)ws_size;
  int ci = 0;
  const float* x     = (const float*)d_in[ci++];
  const int*   masks = (const int*)d_in[ci++];
  auto nextf = [&]() { return (const float*)d_in[ci++]; };
  auto getbn = [&]() { BNP p; p.g = nextf(); p.b = nextf(); p.m = nextf(); p.v = nextf(); return p; };

  const float* stem_w  = nextf();
  BNP          stem_bn = getbn();

  BlkP blks[4][2];
  for (int li = 0; li < 4; ++li)
    for (int bi = 0; bi < 2; ++bi) {
      BlkP bp;
      bp.w1 = nextf(); bp.bn1 = getbn();
      bp.w2 = nextf(); bp.bn2 = getbn();
      bp.has_down = (bi == 0 && li > 0);
      bp.dw = nullptr;
      if (bp.has_down) { bp.dw = nextf(); bp.dbn = getbn(); }
      blks[li][bi] = bp;
    }
  FcP fc[3];
  for (int i = 0; i < 3; ++i) { fc[i].w = nextf(); fc[i].b = nextf(); fc[i].bn = getbn(); }

  // ---- workspace bump allocator (256B aligned) ----
  uint8_t* wbase = (uint8_t*)d_ws;
  size_t off = 0;
  auto alloc = [&](size_t bytes) -> void* {
    off = (off + 255) & ~(size_t)255;
    void* p = wbase + off;
    off += bytes;
    return p;
  };

  auto launch_repack = [&](const float* src, int O, int I, int KH, int KW,
                           int Kpad) -> bf16_t* {
    int Ktot = I * KH * KW;
    bf16_t* dst = (bf16_t*)alloc((size_t)O * Kpad * sizeof(bf16_t));
    int total = O * Kpad;
    repack_w<<<(total + 255) / 256, 256, 0, stream>>>(src, dst, I, KH, KW,
                                                      Ktot, Kpad, total);
    return dst;
  };
  auto launch_bn = [&](BNP bn, const float* bias, int C, float** sc, float** sh) {
    *sc = (float*)alloc((size_t)C * sizeof(float));
    *sh = (float*)alloc((size_t)C * sizeof(float));
    bn_fold<<<(C + 63) / 64, 64, 0, stream>>>(bn.g, bn.b, bn.m, bn.v, bias, *sc, *sh, C);
  };
  auto launch_conv = [&](const bf16_t* in, const bf16_t* wt, bf16_t* out,
                         const float* sc, const float* sh, const bf16_t* res,
                         int Nimg, int H, int W, int Cin, int Cout,
                         int KH, int KW, int stride, int pad, int Kpad, int flags) {
    int Ho = (H + 2 * pad - KH) / stride + 1;
    int Wo = (W + 2 * pad - KW) / stride + 1;
    int M = Nimg * Ho * Wo;
    int Ktot = Cin * KH * KW;
    dim3 grid((M + BM - 1) / BM, Cout / BN);
    gemm_conv_bf16_wmma<<<grid, 256, 0, stream>>>(in, wt, out, sc, sh, res,
        H, W, Cin, Ho, Wo, Cout, KH, KW, stride, pad, Ktot, Kpad, M, flags);
  };

  const int NIMG = 64;
  bf16_t* stem_in  = (bf16_t*)alloc((size_t)NIMG * 256 * 256 * 3 * sizeof(bf16_t));
  bf16_t* stem_out = (bf16_t*)alloc((size_t)NIMG * 128 * 128 * 64 * sizeof(bf16_t));
  const size_t bigsz = (size_t)NIMG * 64 * 64 * 64 * sizeof(bf16_t);
  bf16_t* bufs[3] = { (bf16_t*)alloc(bigsz), (bf16_t*)alloc(bigsz), (bf16_t*)alloc(bigsz) };

  // ---- stem: mask-mul -> 7x7 s2 conv (implicit GEMM, K padded 147->160) ----
  {
    int total = NIMG * 256 * 256 * 3;
    stem_prep<<<(total + 255) / 256, 256, 0, stream>>>(x, masks, stem_in, total);
  }
  bf16_t* stem_wp = launch_repack(stem_w, 64, 3, 7, 7, 160);
  float *ssc, *ssh; launch_bn(stem_bn, nullptr, 64, &ssc, &ssh);
  launch_conv(stem_in, stem_wp, stem_out, ssc, ssh, nullptr,
              NIMG, 256, 256, 3, 64, 7, 7, 2, 3, 160, /*relu*/1);
  {
    int total = NIMG * 64 * 64 * 64;
    maxpool3x3s2<<<(total + 255) / 256, 256, 0, stream>>>(stem_out, bufs[0],
                                                          128, 128, 64, 64, 64, total);
  }

  // ---- residual layers ----
  const int widths[4]  = {64, 128, 256, 512};
  const int lstride[4] = {1, 2, 2, 2};
  int H = 64, Cin = 64;
  bf16_t* cur = bufs[0];
  for (int li = 0; li < 4; ++li) {
    int cout = widths[li];
    for (int bi = 0; bi < 2; ++bi) {
      BlkP& bp = blks[li][bi];
      int s = (bi == 0) ? lstride[li] : 1;
      int cin_b = (bi == 0) ? Cin : cout;
      int Ho = (H + 2 - 3) / s + 1;
      bf16_t *tmp = nullptr, *other = nullptr;
      for (int q = 0; q < 3; ++q)
        if (bufs[q] != cur) { if (!tmp) tmp = bufs[q]; else other = bufs[q]; }

      // conv1 + bn1 + relu : cur -> tmp
      bf16_t* w1p = launch_repack(bp.w1, cout, cin_b, 3, 3, cin_b * 9);
      float *sc1, *sh1; launch_bn(bp.bn1, nullptr, cout, &sc1, &sh1);
      launch_conv(cur, w1p, tmp, sc1, sh1, nullptr,
                  NIMG, H, H, cin_b, cout, 3, 3, s, 1, cin_b * 9, 1);

      // identity path
      const bf16_t* idn; bf16_t* outb;
      if (bp.has_down) {
        bf16_t* dwp = launch_repack(bp.dw, cout, cin_b, 1, 1, cin_b);
        float *dsc, *dsh; launch_bn(bp.dbn, nullptr, cout, &dsc, &dsh);
        launch_conv(cur, dwp, other, dsc, dsh, nullptr,
                    NIMG, H, H, cin_b, cout, 1, 1, s, 0, cin_b, 0);
        idn = other; outb = cur;              // cur is dead after conv1/down
      } else {
        idn = cur; outb = other;
      }

      // conv2 + bn2 + residual + relu : tmp -> outb
      bf16_t* w2p = launch_repack(bp.w2, cout, cout, 3, 3, cout * 9);
      float *sc2, *sh2; launch_bn(bp.bn2, nullptr, cout, &sc2, &sh2);
      launch_conv(tmp, w2p, outb, sc2, sh2, idn,
                  NIMG, Ho, Ho, cout, cout, 3, 3, 1, 1, cout * 9, 1);
      cur = outb;
      H = Ho;
    }
    Cin = cout;
  }

  // ---- global avg pool -> FC head (WMMA GEMMs, bias+BN folded, GELU) ----
  bf16_t* pooled = (bf16_t*)alloc((size_t)NIMG * 512 * sizeof(bf16_t));
  avgpool_global<<<(NIMG * 512 + 255) / 256, 256, 0, stream>>>(cur, pooled,
                                                               64, 512, NIMG * 512);
  const int fdim[4] = {512, 256, 128, 64};
  bf16_t* fin = pooled;
  for (int i = 0; i < 3; ++i) {
    bf16_t* fbuf = (bf16_t*)alloc((size_t)NIMG * fdim[i + 1] * sizeof(bf16_t));
    bf16_t* wp = launch_repack(fc[i].w, fdim[i + 1], fdim[i], 1, 1, fdim[i]);
    float *sc, *sh; launch_bn(fc[i].bn, fc[i].b, fdim[i + 1], &sc, &sh);
    launch_conv(fin, wp, fbuf, sc, sh, nullptr,
                NIMG, 1, 1, fdim[i], fdim[i + 1], 1, 1, 1, 0, fdim[i], /*gelu*/2);
    fin = fbuf;
  }

  final_mean<<<1, 128, 0, stream>>>(fin, (float*)d_out, 128);
}
